// Model_42769284334197
// MI455X (gfx1250) — matrix operand order repaired and checked by reading, not compiled, not run on hardware
//
#include <hip/hip_runtime.h>
#include <hip/hip_bf16.h>

#define H 64

typedef __attribute__((ext_vector_type(16))) _Float16 v16h;
typedef __attribute__((ext_vector_type(8)))  float    v8f;

// ---------------------------------------------------------------------------
// A-tile loader: 16-bit A-matrix 16x32 layout (CDNA5 ISA 7.12.2)
//   lane(0..15)=row M, hi=lane>>4 selects K-halves; element i -> K = (i&7) + 2*(i&8) + 8*hi
// Converts f32 row (optionally scaled, for msum/cnt mean) to f16 on the fly.
__device__ __forceinline__ v16h load_A_f16(const float* __restrict__ rowp, int kk,
                                           int hi, float scale) {
  v16h a;
#pragma unroll
  for (int i = 0; i < 16; ++i) {
    const int k = (i & 7) + ((i & 8) << 1) + 8 * hi;
    a[i] = (_Float16)(rowp[kk + k] * scale);
  }
  return a;
}

// B-tile loader: 16-bit B-matrix 32x16 layout; lane(0..15)=col N, element i -> K = (i&7) + 2*(i&8) + 8*hi
__device__ __forceinline__ v16h load_B_f16(const float* __restrict__ w, int kk, int c0,
                                           int lane) {
  v16h b;
  const int hi = lane >> 4;
  const float* wp = w + (size_t)kk * H + c0 + (lane & 15);
#pragma unroll
  for (int i = 0; i < 16; ++i) {
    b[i] = (_Float16)wp[(size_t)((i & 7) + ((i & 8) << 1) + 8 * hi) * H];
  }
  return b;
}

// out[row] = act( (msum[row]/max(cnt,1)) @ wl + xdst[row] @ wr + bias )
// One wave32 per 16 output rows; K=64 split into two 16x16x32 WMMA steps per source.
template <bool RELU>
__global__ void sage_wmma_kernel(const float* __restrict__ msum,
                                 const int*   __restrict__ cnt,
                                 const float* __restrict__ xdst,
                                 const float* __restrict__ wl,
                                 const float* __restrict__ wr,
                                 const float* __restrict__ bias,
                                 float* __restrict__ out, int n) {
  const int lane = threadIdx.x & 31;
  const int wave = (int)((blockIdx.x * blockDim.x + threadIdx.x) >> 5);
  const int row0 = wave * 16;
  if (row0 >= n) return;                 // wave-uniform: EXEC stays all-ones for WMMA
  const int ml = lane & 15;
  const int hi = lane >> 4;

  const int arow = row0 + ml;
  const float inv = 1.0f / fmaxf((float)cnt[arow], 1.0f);
  const float* mrow = msum + (size_t)arow * H;
  const float* xrow = xdst + (size_t)arow * H;

  const v16h am0 = load_A_f16(mrow, 0,  hi, inv);
  const v16h am1 = load_A_f16(mrow, 32, hi, inv);
  const v16h ax0 = load_A_f16(xrow, 0,  hi, 1.0f);
  const v16h ax1 = load_A_f16(xrow, 32, hi, 1.0f);

#pragma unroll
  for (int c0 = 0; c0 < H; c0 += 16) {
    const float bv = bias[c0 + ml];
    v8f c;
#pragma unroll
    for (int r = 0; r < 8; ++r) c[r] = bv;

    const v16h bl0 = load_B_f16(wl, 0,  c0, lane);
    const v16h bl1 = load_B_f16(wl, 32, c0, lane);
    const v16h br0 = load_B_f16(wr, 0,  c0, lane);
    const v16h br1 = load_B_f16(wr, 32, c0, lane);

    c = __builtin_amdgcn_wmma_f32_16x16x32_f16(false, am0, false, bl0, (short)0, c, false, false);
    c = __builtin_amdgcn_wmma_f32_16x16x32_f16(false, am1, false, bl1, (short)0, c, false, false);
    c = __builtin_amdgcn_wmma_f32_16x16x32_f16(false, ax0, false, br0, (short)0, c, false, false);
    c = __builtin_amdgcn_wmma_f32_16x16x32_f16(false, ax1, false, br1, (short)0, c, false, false);

#pragma unroll
    for (int r = 0; r < 8; ++r) {
      float v = c[r];
      if (RELU) v = fmaxf(v, 0.0f);
      out[(size_t)(row0 + r + 8 * hi) * H + c0 + ml] = v;
    }
  }
}

// In-degree counts for both edge directions (1 thread per edge, native u32 atomics).
__global__ void count_kernel(const int* __restrict__ src, const int* __restrict__ dst,
                             int* __restrict__ cg, int* __restrict__ cd, int ne) {
  const int e = blockIdx.x * blockDim.x + threadIdx.x;
  if (e >= ne) return;
  atomicAdd(&cd[dst[e]], 1);
  atomicAdd(&cg[src[e]], 1);
}

// acc[dst[e]] += x[src[e]]  (one wave per edge, float2 per lane, native f32 atomics)
__global__ void scatter_add_kernel(const float* __restrict__ x,
                                   const int* __restrict__ src,
                                   const int* __restrict__ dst,
                                   float* __restrict__ acc, int ne) {
  const long long t = (long long)blockIdx.x * blockDim.x + threadIdx.x;
  const int e = (int)(t >> 5);
  const int lane = (int)(t & 31);
  if (e >= ne) return;
  const int s = src[e];
  const int d = dst[e];
  const float2 v = *(const float2*)(x + (size_t)s * H + lane * 2);
  float* p = acc + (size_t)d * H + lane * 2;
  unsafeAtomicAdd(p,     v.x);   // global_atomic_add_f32
  unsafeAtomicAdd(p + 1, v.y);
}

// out[l] = dot(hg[ls[l]], hd[ld[l]]) over H=64; one wave per label edge.
__global__ void dot_kernel(const float* __restrict__ hg, const float* __restrict__ hd,
                           const int* __restrict__ ls, const int* __restrict__ ld,
                           float* __restrict__ out, int nl) {
  const long long t = (long long)blockIdx.x * blockDim.x + threadIdx.x;
  const int e = (int)(t >> 5);
  const int lane = (int)(t & 31);
  if (e >= nl) return;
  const float2 a = *(const float2*)(hg + (size_t)ls[e] * H + lane * 2);
  const float2 b = *(const float2*)(hd + (size_t)ld[e] * H + lane * 2);
  float v = a.x * b.x + a.y * b.y;
#pragma unroll
  for (int off = 16; off > 0; off >>= 1) v += __shfl_xor(v, off, 32);
  if (lane == 0) out[e] = v;
}

extern "C" void kernel_launch(void* const* d_in, const int* in_sizes, int n_in,
                              void* d_out, int out_size, void* d_ws, size_t ws_size,
                              hipStream_t stream) {
  const float* gene_emb    = (const float*)d_in[0];
  const float* disease_emb = (const float*)d_in[1];
  const float* w1_gd_l = (const float*)d_in[2];
  const float* w1_gd_r = (const float*)d_in[3];
  const float* w1_dg_l = (const float*)d_in[4];
  const float* w1_dg_r = (const float*)d_in[5];
  const float* w2_gd_l = (const float*)d_in[6];
  const float* w2_gd_r = (const float*)d_in[7];
  const float* w2_dg_l = (const float*)d_in[8];
  const float* w2_dg_r = (const float*)d_in[9];
  const float* b1_gd = (const float*)d_in[10];
  const float* b1_dg = (const float*)d_in[11];
  const float* b2_gd = (const float*)d_in[12];
  const float* b2_dg = (const float*)d_in[13];
  const int* edge_src  = (const int*)d_in[14];
  const int* edge_dst  = (const int*)d_in[15];
  const int* label_src = (const int*)d_in[16];
  const int* label_dst = (const int*)d_in[17];

  const int NG = in_sizes[0] / H;
  const int ND = in_sizes[1] / H;
  const int NE = in_sizes[14];
  const int NL = in_sizes[16];

  // Workspace layout
  float* msum_d = (float*)d_ws;                    // ND*H  (also reused for layer 2)
  float* msum_g = msum_d + (size_t)ND * H;         // NG*H
  float* h_d    = msum_g + (size_t)NG * H;         // ND*H
  float* h_g    = h_d    + (size_t)ND * H;         // NG*H
  float* h_d2   = h_g    + (size_t)NG * H;         // ND*H
  float* h_g2   = h_d2   + (size_t)ND * H;         // NG*H
  int*   cnt_d  = (int*)(h_g2 + (size_t)NG * H);   // ND
  int*   cnt_g  = cnt_d + ND;                      // NG

  const size_t msum_bytes = (size_t)(ND + NG) * H * sizeof(float);

  const int edge_blk = (int)(((long long)NE * 32 + 255) / 256);
  const int cnt_blk  = (NE + 255) / 256;
  const int gdim_d   = (ND + 127) / 128;           // 256 thr = 8 waves = 128 rows
  const int gdim_g   = (NG + 127) / 128;
  const int dot_blk  = (int)(((long long)NL * 32 + 255) / 256);

  // ---- Layer 1 ----
  hipMemsetAsync(msum_d, 0, msum_bytes, stream);                     // msum_d + msum_g
  hipMemsetAsync(cnt_d, 0, (size_t)(ND + NG) * sizeof(int), stream); // cnt_d + cnt_g
  count_kernel<<<cnt_blk, 256, 0, stream>>>(edge_src, edge_dst, cnt_g, cnt_d, NE);
  scatter_add_kernel<<<edge_blk, 256, 0, stream>>>(gene_emb,    edge_src, edge_dst, msum_d, NE);
  scatter_add_kernel<<<edge_blk, 256, 0, stream>>>(disease_emb, edge_dst, edge_src, msum_g, NE);
  sage_wmma_kernel<true><<<gdim_d, 256, 0, stream>>>(msum_d, cnt_d, disease_emb,
                                                     w1_gd_l, w1_gd_r, b1_gd, h_d, ND);
  sage_wmma_kernel<true><<<gdim_g, 256, 0, stream>>>(msum_g, cnt_g, gene_emb,
                                                     w1_dg_l, w1_dg_r, b1_dg, h_g, NG);

  // ---- Layer 2 ----
  hipMemsetAsync(msum_d, 0, msum_bytes, stream);
  scatter_add_kernel<<<edge_blk, 256, 0, stream>>>(h_g, edge_src, edge_dst, msum_d, NE);
  scatter_add_kernel<<<edge_blk, 256, 0, stream>>>(h_d, edge_dst, edge_src, msum_g, NE);
  sage_wmma_kernel<false><<<gdim_d, 256, 0, stream>>>(msum_d, cnt_d, h_d,
                                                      w2_gd_l, w2_gd_r, b2_gd, h_d2, ND);
  sage_wmma_kernel<false><<<gdim_g, 256, 0, stream>>>(msum_g, cnt_g, h_g,
                                                      w2_dg_l, w2_dg_r, b2_dg, h_g2, NG);

  // ---- Classifier ----
  dot_kernel<<<dot_blk, 256, 0, stream>>>(h_g2, h_d2, label_src, label_dst,
                                          (float*)d_out, NL);
}
